// TaskRouter_28063316312419
// MI455X (gfx1250) — matrix operand order repaired and checked by reading, not compiled, not run on hardware
//
#include <hip/hip_runtime.h>
#include <hip/hip_bf16.h>

#define BATCH    262144
#define D_MODEL  256
#define HIDDEN   128
#define NUM_TASKS 3

#define BM        128   // batch rows per workgroup
#define A_STRIDE  264   // ushort stride: 528B rows -> 16B aligned, bank-shift 4
#define W_STRIDE  264
#define H_STRIDE  132   // float stride: same 528B rows (aliases A tile)
#define W1_BYTES  (HIDDEN * W_STRIDE * 2)   // 67584

typedef __bf16 bf16x2 __attribute__((ext_vector_type(2)));
typedef __bf16 bf16x8 __attribute__((ext_vector_type(8)));
typedef __bf16 v16bf  __attribute__((ext_vector_type(16)));
typedef float  v8f    __attribute__((ext_vector_type(8)));

static __device__ __forceinline__ unsigned short f32_to_bf16(float x) {
    unsigned int u = __float_as_uint(x);
    u += 0x7FFFu + ((u >> 16) & 1u);          // round-to-nearest-even
    return (unsigned short)(u >> 16);
}

// pack two f32 -> {lo=x, hi=y} bf16 pair; prefer v_cvt_pk_bf16_f32 when available
static __device__ __forceinline__ unsigned int pack_bf16(float x, float y) {
#if __has_builtin(__builtin_amdgcn_cvt_pk_bf16_f32)
    bf16x2 p = __builtin_amdgcn_cvt_pk_bf16_f32(x, y);
    return __builtin_bit_cast(unsigned int, p);
#else
    return ((unsigned)f32_to_bf16(y) << 16) | f32_to_bf16(x);
#endif
}

static __device__ __forceinline__ v16bf cat8(bf16x8 lo, bf16x8 hi) {
    v16bf v;
#pragma unroll
    for (int i = 0; i < 8; ++i) { v[i] = lo[i]; v[i + 8] = hi[i]; }
    return v;
}

// ---- one-shot pre-pass: W1[256][128] f32 -> padded W1^T [128][W_STRIDE] bf16 in d_ws ----
__global__ __launch_bounds__(256) void pack_w1_kernel(const float* __restrict__ W1,
                                                      unsigned short* __restrict__ w1bf)
{
    int base = (blockIdx.x * 256 + threadIdx.x) * 4;     // flat into [256][128]
    if (base < D_MODEL * HIDDEN) {
        float4 w = *(const float4*)(W1 + base);
        int k = base >> 7;                               // 0..255
        int n = base & 127;                              // multiple of 4
        w1bf[(n + 0) * W_STRIDE + k] = f32_to_bf16(w.x);
        w1bf[(n + 1) * W_STRIDE + k] = f32_to_bf16(w.y);
        w1bf[(n + 2) * W_STRIDE + k] = f32_to_bf16(w.z);
        w1bf[(n + 3) * W_STRIDE + k] = f32_to_bf16(w.w);
    }
}

__global__ __launch_bounds__(256) void task_router_kernel(
    const float* __restrict__ h_t, const unsigned short* __restrict__ w1bf,
    const float* __restrict__ b1,  const float* __restrict__ W2,
    const float* __restrict__ b2,  const float* __restrict__ cb,
    float* __restrict__ e_out,     float* __restrict__ g_out)
{
    __shared__ unsigned short s_w1[HIDDEN * W_STRIDE];   // W1^T, bf16 (async-filled)
    __shared__ float          s_hid[BM * H_STRIDE];      // phase1: bf16 A tile (alias); phase2: f32 hidden
    __shared__ float          s_w2[HIDDEN * 4];
    __shared__ float          s_cb[NUM_TASKS * D_MODEL];
    __shared__ float          s_b1[HIDDEN];
    __shared__ float          s_b2[4];
    __shared__ float          s_rw0[BM], s_rw1[BM];
    __shared__ int            s_ri0[BM], s_ri1[BM];

    unsigned short* s_a = (unsigned short*)s_hid;

    const int    tid     = threadIdx.x;
    const size_t rowBase = (size_t)blockIdx.x * BM;

    // ---- async global->LDS copy of prepacked W1^T (CDNA5 ASYNCcnt path) ----
    {
        // low 32 bits of a shared pointer == LDS byte offset (aperture mapping)
        unsigned ldsBase = (unsigned)(size_t)(void*)&s_w1[0];
#pragma unroll
        for (int off = 0; off < W1_BYTES; off += 256 * 16) {
            int myOff = off + tid * 16;
            if (myOff < W1_BYTES) {
                unsigned loff = ldsBase + (unsigned)myOff;
                unsigned goff = (unsigned)myOff;
                asm volatile("global_load_async_to_lds_b128 %0, %1, %2"
                             :: "v"(loff), "v"(goff), "s"(w1bf)
                             : "memory");
            }
        }
    }

    // Warm L2 with this block's h_t tile.
    __builtin_prefetch(h_t + (rowBase + (size_t)(tid >> 1)) * D_MODEL + (tid & 1) * 128, 0, 3);

    // ---- Stage h_t tile (128x256 f32 -> bf16 row-major, padded) ----
#pragma unroll
    for (int i = 0; i < 32; ++i) {
        int chunk = tid + i * 256;                       // float4 chunk id
        int r = chunk >> 6;                              // 0..127
        int c = (chunk & 63) * 4;                        // 0..252
        float4 hv = *(const float4*)(h_t + (rowBase + r) * D_MODEL + c);
        *(unsigned int*)&s_a[r * A_STRIDE + c]     = pack_bf16(hv.x, hv.y);
        *(unsigned int*)&s_a[r * A_STRIDE + c + 2] = pack_bf16(hv.z, hv.w);
    }

    // ---- Small operands ----
    if (tid < HIDDEN) s_b1[tid] = b1[tid];
    if (tid < NUM_TASKS) s_b2[tid] = b2[tid];
    for (int idx = tid; idx < HIDDEN * NUM_TASKS; idx += 256)
        s_w2[(idx / 3) * 4 + (idx % 3)] = W2[idx];
    for (int idx = tid; idx < NUM_TASKS * D_MODEL; idx += 256)
        s_cb[idx] = cb[idx];

    asm volatile("s_wait_asynccnt 0x0" ::: "memory");    // W1 tile resident in LDS
    __syncthreads();

    // ---- GEMM1: wave w computes rows [w*16, w*16+16) x all 128 hidden ----
    const int wave = tid >> 5;
    const int lane = tid & 31;
    const int half = lane >> 4;   // 0: K 0..7/16..23 (A and B); 1: shifted halves
    const int l16  = lane & 15;

    v8f acc[8] = {};
    const unsigned short* aRow = s_a + (wave * 16 + l16) * A_STRIDE + half * 8;

#pragma unroll 2
    for (int kb = 0; kb < D_MODEL; kb += 32) {
        // A fragment (16x32 bf16): lane holds row M=l16, K = kb + {0..7,16..23} (+8 for upper half)
        v16bf a = cat8(*(const bf16x8*)(aRow + kb),
                       *(const bf16x8*)(aRow + kb + 16));
#pragma unroll
        for (int nt = 0; nt < 8; ++nt) {
            // B fragment (32x16 bf16): lane holds col N=nt*16+l16, K = kb + {0..7,16..23} (+8 for upper half)
            const unsigned short* bp = s_w1 + (nt * 16 + l16) * W_STRIDE + kb + half * 8;
            v16bf b = cat8(*(const bf16x8*)bp, *(const bf16x8*)(bp + 16));
            acc[nt] = __builtin_amdgcn_wmma_f32_16x16x32_bf16(
                false, a, false, b, (short)0, acc[nt], false, false);
        }
    }
    __syncthreads();   // all A-tile reads done before aliasing it as f32 hidden

    // ---- bias + ReLU, spill hidden to LDS ----
    // C layout: VGPR r, lanes 0-15 -> M=r,N=lane ; lanes 16-31 -> M=8+r,N=lane-16
#pragma unroll
    for (int nt = 0; nt < 8; ++nt) {
        int n = nt * 16 + l16;
        float bias = s_b1[n];
#pragma unroll
        for (int r = 0; r < 8; ++r) {
            int m = wave * 16 + half * 8 + r;
            float h = fmaxf(acc[nt][r] + bias, 0.f);
            s_hid[m * H_STRIDE + n] = h;
        }
    }
    __syncthreads();

    // ---- GEMM2 (N=3) + softmax + top-2 routing, one thread per row ----
    if (tid < BM) {
        const int r = tid;
        const float* hr = &s_hid[r * H_STRIDE];
        float l0 = s_b2[0], l1 = s_b2[1], l2 = s_b2[2];
#pragma unroll 4
        for (int n = 0; n < HIDDEN; ++n) {
            float h = hr[n];
            l0 = fmaf(h, s_w2[n * 4 + 0], l0);
            l1 = fmaf(h, s_w2[n * 4 + 1], l1);
            l2 = fmaf(h, s_w2[n * 4 + 2], l2);
        }
        float mx = fmaxf(l0, fmaxf(l1, l2));
        float e0 = __expf(l0 - mx), e1 = __expf(l1 - mx), e2 = __expf(l2 - mx);
        float inv = 1.f / (e0 + e1 + e2);
        float g0 = e0 * inv, g1 = e1 * inv, g2 = e2 * inv;

        size_t grow = (rowBase + r) * 3;
        g_out[grow + 0] = g0; g_out[grow + 1] = g1; g_out[grow + 2] = g2;

        int i0 = 0; float v0 = g0;
        if (g1 > v0) { v0 = g1; i0 = 1; }
        if (g2 > v0) { v0 = g2; i0 = 2; }
        int i1; float v1;
        if (i0 == 0)      { if (g1 >= g2) { i1 = 1; v1 = g1; } else { i1 = 2; v1 = g2; } }
        else if (i0 == 1) { if (g0 >= g2) { i1 = 0; v1 = g0; } else { i1 = 2; v1 = g2; } }
        else              { if (g0 >= g1) { i1 = 0; v1 = g0; } else { i1 = 1; v1 = g1; } }

        bool use_topk = v0 < 0.7f;
        s_rw0[r] = use_topk ? v0 : 1.0f;
        s_rw1[r] = use_topk ? v1 : 0.0f;
        s_ri0[r] = i0;
        s_ri1[r] = i1;
    }
    __syncthreads();

    // ---- e_task: one coalesced 1KB row store per iteration ----
    const int c = tid;  // 256 threads == 256 cols
#pragma unroll 4
    for (int r = 0; r < BM; ++r) {
        float v = s_rw0[r] * s_cb[s_ri0[r] * D_MODEL + c]
                + s_rw1[r] * s_cb[s_ri1[r] * D_MODEL + c];
        e_out[(rowBase + r) * D_MODEL + c] = v;
    }
}

extern "C" void kernel_launch(void* const* d_in, const int* in_sizes, int n_in,
                              void* d_out, int out_size, void* d_ws, size_t ws_size,
                              hipStream_t stream) {
    const float* h_t = (const float*)d_in[0];
    const float* W1  = (const float*)d_in[1];
    const float* b1  = (const float*)d_in[2];
    const float* W2  = (const float*)d_in[3];
    const float* b2  = (const float*)d_in[4];
    const float* cb  = (const float*)d_in[5];
    float* e_out = (float*)d_out;
    float* g_out = e_out + (size_t)BATCH * D_MODEL;
    unsigned short* w1bf = (unsigned short*)d_ws;        // 67584 B of scratch

    // 1) pack W1^T to bf16 once per launch (tiny; same result every call)
    pack_w1_kernel<<<dim3((D_MODEL * HIDDEN / 4 + 255) / 256), dim3(256), 0, stream>>>(W1, w1bf);
    // 2) fused router kernel
    task_router_kernel<<<dim3(BATCH / BM), dim3(256), 0, stream>>>(
        h_t, w1bf, b1, W2, b2, cb, e_out, g_out);
}
